// WavFeat_42253888258225
// MI455X (gfx1250) — matrix operand order repaired and checked by reading, not compiled, not run on hardware
//
#include <hip/hip_runtime.h>
#include <hip/hip_bf16.h>

// ---------------- problem constants ----------------
#define N_FFTC  2048
#define HOPC    441
#define DIMC    512
#define SIG_LEN 1323000
#define PADC    1024
#define T_FR    3001                    // frames per (b,c) row
#define BCROWS  8                       // b*c
#define M_REAL  (BCROWS * T_FR)         // 24008 frames total
#define M_TILES 1501                    // ceil(M_REAL / 16)
#define M_PADR  (M_TILES * 16)          // 24016 (padded rows)
#define L2C     (N_FFTC + (T_FR - 1) * HOPC)   // 1325048

#define ASTR1   (N_FFTC + 8)            // LDS row stride (bank-conflict pad), encode
#define ASTR2   (DIMC + 8)              // LDS row stride, decode

typedef __attribute__((ext_vector_type(16))) __bf16       v16bf;
typedef __attribute__((ext_vector_type(8)))  float        v8f;
typedef __attribute__((ext_vector_type(4)))  unsigned int v4u;

union Frag {
    v16bf v;
    v4u   q[2];
};

#define WMMA_BF16(acc, a, b) \
    (acc) = __builtin_amdgcn_wmma_f32_16x16x32_bf16(false, (a).v, false, (b).v, (short)0, (acc), false, false)

// ---------------- helper kernels ----------------

// Hann window (periodic): 0.5*(1 - cos(2*pi*n/N))
__global__ void k_window(float* __restrict__ win) {
    int j = blockIdx.x * blockDim.x + threadIdx.x;
    if (j < N_FFTC) {
        float ph = 6.283185307179586f * (float)j * (1.0f / (float)N_FFTC);
        win[j] = 0.5f * (1.0f - __cosf(ph));
    }
}

__global__ void k_cvt_bf16(const float* __restrict__ in, __bf16* __restrict__ out, int n) {
    int i = blockIdx.x * blockDim.x + threadIdx.x;
    if (i < n) out[i] = (__bf16)in[i];
}

// Build windowed frames in bf16 with reflect padding; zero-fill padded rows.
__global__ void k_frames(const float* __restrict__ x, const float* __restrict__ win,
                         __bf16* __restrict__ frames) {
    int idx = blockIdx.x * blockDim.x + threadIdx.x;   // < M_PADR * N_FFTC
    int fi = idx >> 11;            // frame index
    int j  = idx & (N_FFTC - 1);   // sample in frame
    __bf16 o = (__bf16)0.0f;
    if (fi < M_REAL) {
        int bc = fi / T_FR;
        int ti = fi - bc * T_FR;
        int s = ti * HOPC + j - PADC;            // source index into x row
        s = (s < 0) ? -s : ((s >= SIG_LEN) ? (2 * SIG_LEN - 2 - s) : s);
        o = (__bf16)(x[(size_t)bc * SIG_LEN + s] * win[j]);
    }
    frames[idx] = o;
}

__global__ void k_zero(float* __restrict__ p, int n) {
    int i = blockIdx.x * blockDim.x + threadIdx.x;
    if (i < n) p[i] = 0.0f;
}

// Analytic fold of the window: win_norm[pos] = sum_t win[pos - t*HOP]
__global__ void k_winnorm(const float* __restrict__ win, float* __restrict__ wn) {
    int pos = blockIdx.x * blockDim.x + threadIdx.x;
    if (pos >= L2C) return;
    int t0 = (pos >= N_FFTC) ? (pos - (N_FFTC - 1) + HOPC - 1) / HOPC : 0;
    int t1 = pos / HOPC; if (t1 > T_FR - 1) t1 = T_FR - 1;
    float s = 0.0f;
    for (int t = t0; t <= t1; ++t) s += win[pos - t * HOPC];
    wn[pos] = s;
}

// ---------------- encode GEMM: feat = frames @ W_enc^T ----------------
// One 16-row M tile per block; A tile staged once in LDS (64 KB) and shared by 8
// waves that each compute a 16x64 strip of the 512-wide output. B fragments are
// double-buffered so global loads overlap the WMMAs.
__global__ __launch_bounds__(256) void k_gemm_enc(const __bf16* __restrict__ A,
                                                  const __bf16* __restrict__ B,
                                                  __bf16* __restrict__ C) {
    __shared__ __bf16 sa[16 * ASTR1];
    const int tid  = threadIdx.x;
    const int lane = tid & 31;
    const int wid  = tid >> 5;
    const int mt   = blockIdx.x;                // grid.x == M_TILES exactly
    const int g  = lane >> 4;
    const int ln = lane & 15;
    const int nbase = wid * 64;                 // 8 waves cover N = 0..511

    // cooperative stage: 16 x 2048 A tile -> LDS (padded row stride)
    {
        const __bf16* ag = A + (size_t)mt * 16 * N_FFTC;
        #pragma unroll
        for (int i = 0; i < 16; ++i) {          // 4096 b128 vectors / 256 threads
            int v = tid + i * 256;
            int r = v >> 8;                     // 2048/8 = 256 vectors per row
            int c = (v & 255) * 8;
            *(v4u*)&sa[r * ASTR1 + c] = *(const v4u*)&ag[(size_t)r * N_FFTC + c];
        }
    }
    __syncthreads();

    const __bf16* bp = B + (size_t)(nbase + ln) * N_FFTC + g * 16;

    v8f c0 = {}, c1 = {}, c2 = {}, c3 = {};
    Frag a0, a1, b0[4], b1[4];

    #pragma unroll
    for (int t = 0; t < 4; ++t) {               // prologue: K-step 0
        b0[t].q[0] = *(const v4u*)(bp + t * 16 * N_FFTC);
        b0[t].q[1] = *(const v4u*)(bp + t * 16 * N_FFTC + 8);
    }
    for (int ks = 0; ks < N_FFTC / 32; ks += 2) {
        #pragma unroll
        for (int t = 0; t < 4; ++t) {           // issue K-step ks+1
            b1[t].q[0] = *(const v4u*)(bp + 32 + t * 16 * N_FFTC);
            b1[t].q[1] = *(const v4u*)(bp + 32 + t * 16 * N_FFTC + 8);
        }
        a0.q[0] = *(const v4u*)&sa[ln * ASTR1 + g * 16 + ks * 32];
        a0.q[1] = *(const v4u*)&sa[ln * ASTR1 + g * 16 + ks * 32 + 8];
        WMMA_BF16(c0, a0, b0[0]);
        WMMA_BF16(c1, a0, b0[1]);
        WMMA_BF16(c2, a0, b0[2]);
        WMMA_BF16(c3, a0, b0[3]);
        #pragma unroll
        for (int t = 0; t < 4; ++t) {           // issue K-step ks+2 (tail overfetch: pads)
            b0[t].q[0] = *(const v4u*)(bp + 64 + t * 16 * N_FFTC);
            b0[t].q[1] = *(const v4u*)(bp + 64 + t * 16 * N_FFTC + 8);
        }
        a1.q[0] = *(const v4u*)&sa[ln * ASTR1 + g * 16 + ks * 32 + 32];
        a1.q[1] = *(const v4u*)&sa[ln * ASTR1 + g * 16 + ks * 32 + 40];
        WMMA_BF16(c0, a1, b1[0]);
        WMMA_BF16(c1, a1, b1[1]);
        WMMA_BF16(c2, a1, b1[2]);
        WMMA_BF16(c3, a1, b1[3]);
        bp += 64;
    }

    __bf16* cp = C + (size_t)(mt * 16 + g * 8) * DIMC + nbase + ln;
    #pragma unroll
    for (int r = 0; r < 8; ++r) {
        cp[(size_t)r * DIMC + 0]  = (__bf16)c0[r];
        cp[(size_t)r * DIMC + 16] = (__bf16)c1[r];
        cp[(size_t)r * DIMC + 32] = (__bf16)c2[r];
        cp[(size_t)r * DIMC + 48] = (__bf16)c3[r];
    }
}

// ---------------- decode GEMM + fused overlap-add ----------------
// A: feat (M_PADR x 512) bf16 staged in LDS (16 KB tile). B: W_dec (2048 x 512) bf16.
// Epilogue scatters each rec element into the fold accumulator with f32 atomics.
__global__ __launch_bounds__(256) void k_gemm_dec(const __bf16* __restrict__ A,
                                                  const __bf16* __restrict__ B,
                                                  float* __restrict__ acc) {
    __shared__ __bf16 sa[16 * ASTR2];
    const int tid  = threadIdx.x;
    const int lane = tid & 31;
    const int wid  = tid >> 5;
    const int mt   = blockIdx.x;
    const int g  = lane >> 4;
    const int ln = lane & 15;
    const int nbase = blockIdx.y * 512 + wid * 64;   // output sample base (0..2047)

    {
        const __bf16* ag = A + (size_t)mt * 16 * DIMC;
        #pragma unroll
        for (int i = 0; i < 4; ++i) {           // 1024 b128 vectors / 256 threads
            int v = tid + i * 256;
            int r = v >> 6;                     // 512/8 = 64 vectors per row
            int c = (v & 63) * 8;
            *(v4u*)&sa[r * ASTR2 + c] = *(const v4u*)&ag[(size_t)r * DIMC + c];
        }
    }
    __syncthreads();

    const __bf16* bp = B + (size_t)(nbase + ln) * DIMC + g * 16;

    v8f c0 = {}, c1 = {}, c2 = {}, c3 = {};
    Frag a0, a1, b0[4], b1[4];

    #pragma unroll
    for (int t = 0; t < 4; ++t) {
        b0[t].q[0] = *(const v4u*)(bp + t * 16 * DIMC);
        b0[t].q[1] = *(const v4u*)(bp + t * 16 * DIMC + 8);
    }
    for (int ks = 0; ks < DIMC / 32; ks += 2) {
        #pragma unroll
        for (int t = 0; t < 4; ++t) {
            b1[t].q[0] = *(const v4u*)(bp + 32 + t * 16 * DIMC);
            b1[t].q[1] = *(const v4u*)(bp + 32 + t * 16 * DIMC + 8);
        }
        a0.q[0] = *(const v4u*)&sa[ln * ASTR2 + g * 16 + ks * 32];
        a0.q[1] = *(const v4u*)&sa[ln * ASTR2 + g * 16 + ks * 32 + 8];
        WMMA_BF16(c0, a0, b0[0]);
        WMMA_BF16(c1, a0, b0[1]);
        WMMA_BF16(c2, a0, b0[2]);
        WMMA_BF16(c3, a0, b0[3]);
        #pragma unroll
        for (int t = 0; t < 4; ++t) {           // tail overfetch lands in padded ws
            b0[t].q[0] = *(const v4u*)(bp + 64 + t * 16 * DIMC);
            b0[t].q[1] = *(const v4u*)(bp + 64 + t * 16 * DIMC + 8);
        }
        a1.q[0] = *(const v4u*)&sa[ln * ASTR2 + g * 16 + ks * 32 + 32];
        a1.q[1] = *(const v4u*)&sa[ln * ASTR2 + g * 16 + ks * 32 + 40];
        WMMA_BF16(c0, a1, b1[0]);
        WMMA_BF16(c1, a1, b1[1]);
        WMMA_BF16(c2, a1, b1[2]);
        WMMA_BF16(c3, a1, b1[3]);
        bp += 64;
    }

    const int fi0 = mt * 16 + g * 8;
    #pragma unroll
    for (int r = 0; r < 8; ++r) {
        int fi = fi0 + r;
        if (fi < M_REAL) {                       // drop tile-padding rows
            int bc = fi / T_FR;
            int ti = fi - bc * T_FR;
            float* dst = acc + (size_t)bc * L2C + (size_t)ti * HOPC + nbase + ln;
            atomicAdd(dst + 0,  c0[r]);
            atomicAdd(dst + 16, c1[r]);
            atomicAdd(dst + 32, c2[r]);
            atomicAdd(dst + 48, c3[r]);
        }
    }
}

// ---------------- finalize: normalize + slice ----------------
__global__ void k_final(const float* __restrict__ acc, const float* __restrict__ wn,
                        float* __restrict__ out) {
    int idx = blockIdx.x * blockDim.x + threadIdx.x;
    if (idx >= BCROWS * SIG_LEN) return;
    int bc = idx / SIG_LEN;
    int l  = idx - bc * SIG_LEN;
    float w = wn[PADC + l];
    if (w < 1e-8f) w = 1e-8f;
    out[idx] = acc[(size_t)bc * L2C + PADC + l] / w;
}

// ---------------- launch ----------------
extern "C" void kernel_launch(void* const* d_in, const int* in_sizes, int n_in,
                              void* d_out, int out_size, void* d_ws, size_t ws_size,
                              hipStream_t stream) {
    const float* x    = (const float*)d_in[0];   // (4,2,1323000)
    const float* Wenc = (const float*)d_in[1];   // (512,2048)
    const float* Wdec = (const float*)d_in[2];   // (2048,512)
    float* out = (float*)d_out;

    auto up = [](size_t v) { return (v + 255) & ~(size_t)255; };
    char* ws = (char*)d_ws;
    size_t o = 0;
    float*  win     = (float*)(ws + o);  o += up((size_t)N_FFTC * 4);
    __bf16* wenc_bf = (__bf16*)(ws + o); o += up((size_t)DIMC * N_FFTC * 2 + 256);
    __bf16* wdec_bf = (__bf16*)(ws + o); o += up((size_t)N_FFTC * DIMC * 2 + 256);
    __bf16* frames  = (__bf16*)(ws + o); o += up((size_t)M_PADR * N_FFTC * 2 + 256);
    __bf16* feat    = (__bf16*)(ws + o); o += up((size_t)M_PADR * DIMC * 2 + 256);
    float*  acc     = (float*)(ws + o);  o += up((size_t)BCROWS * L2C * 4);
    float*  wnorm   = (float*)(ws + o);  o += up((size_t)L2C * 4);
    (void)ws_size; (void)in_sizes; (void)n_in; (void)out_size;

    k_window<<<(N_FFTC + 255) / 256, 256, 0, stream>>>(win);

    int nw = DIMC * N_FFTC;
    k_cvt_bf16<<<(nw + 255) / 256, 256, 0, stream>>>(Wenc, wenc_bf, nw);
    k_cvt_bf16<<<(nw + 255) / 256, 256, 0, stream>>>(Wdec, wdec_bf, nw);

    int nfr = M_PADR * N_FFTC;                       // 49,184,768 (multiple of 256)
    k_frames<<<nfr / 256, 256, 0, stream>>>(x, win, frames);

    int nacc = BCROWS * L2C;
    k_zero<<<(nacc + 255) / 256, 256, 0, stream>>>(acc, nacc);
    k_winnorm<<<(L2C + 255) / 256, 256, 0, stream>>>(win, wnorm);

    k_gemm_enc<<<dim3(M_TILES, 1), 256, 0, stream>>>(frames, wenc_bf, feat);

    k_gemm_dec<<<dim3(M_TILES, N_FFTC / 512), 256, 0, stream>>>(feat, wdec_bf, acc);

    int nout = BCROWS * SIG_LEN;
    k_final<<<(nout + 255) / 256, 256, 0, stream>>>(acc, wnorm, out);
}
